// SA_Block_Ancher_38130719654100
// MI455X (gfx1250) — matrix operand order repaired and checked by reading, not compiled, not run on hardware
//
#include <hip/hip_runtime.h>
#include <hip/hip_bf16.h>

typedef __attribute__((ext_vector_type(16))) _Float16 v16h;
typedef __attribute__((ext_vector_type(8)))  _Float16 v8h;
typedef __attribute__((ext_vector_type(8)))  float    v8f;

#define PITCH 72   // LDS row pitch in halves: 144 bytes, multiple of 16

// ---------------------------------------------------------------------------
// Fragment loads from LDS as 2x ds_load_b128 each (ISA 7.12.2 layouts, wave32)
// ---------------------------------------------------------------------------
// A tile in LDS: natural [m][k], pitch PITCH.  Lane fragment:
//   e=0..7  -> k = kbase + hi*8 + e        (contiguous 8 halves)
//   e=8..15 -> k = kbase + 16 + hi*8 + e-8 (contiguous 8 halves)
__device__ inline v16h frag_a(const _Float16* sm, int m0, int kbase, int lane) {
  int ln = lane & 15, hi = lane >> 4;
  const _Float16* p = sm + (m0 + ln) * PITCH + kbase + hi * 8;
  union { v16h v; v8h h[2]; } u;
  u.h[0] = *(const v8h*)p;
  u.h[1] = *(const v8h*)(p + 16);
  return u.v;
}

// B tile in LDS: TRANSPOSED [n][k], pitch PITCH.  Lane fragment:
//   e=0..7  -> k = kbase + hi*8 + e        (contiguous 8 halves)
//   e=8..15 -> k = kbase + 16 + hi*8 + e-8 (contiguous 8 halves)
__device__ inline v16h frag_b(const _Float16* smT, int n0, int kbase, int lane) {
  int ln = lane & 15, hi = lane >> 4;
  const _Float16* p = smT + (n0 + ln) * PITCH + kbase + hi * 8;
  union { v16h v; v8h h[2]; } u;
  u.h[0] = *(const v8h*)p;
  u.h[1] = *(const v8h*)(p + 16);
  return u.v;
}

// ---------------------------------------------------------------------------
// Kernel 1: 1x1 conv as GEMM  Y[b](64xN) = W(64x64) * X[b](64xN) + bias
// mode 0: f32 natural (q) | mode 1: f16 natural (K,V) | mode 2: f16 transposed (Q^T)
// grid: (N/64, B), block 256 = 8 waves, each wave a 16x32 tile (64x64 block tile)
// ---------------------------------------------------------------------------
__global__ void conv_gemm(const float* __restrict__ X, const float* __restrict__ Wt,
                          const float* __restrict__ bias,
                          float* __restrict__ outF, _Float16* __restrict__ outH,
                          int mode, int Ntot) {
  __shared__ __align__(16) _Float16 Ah[64 * PITCH];
  __shared__ __align__(16) _Float16 Bt[64 * PITCH];
  const int b = blockIdx.y;
  const int n_base = blockIdx.x * 64;
  const float* Xb = X + (long)b * 64 * Ntot;
  const int tid = threadIdx.x;

  for (int i = tid; i < 64 * 64; i += 256) {          // W -> LDS (f16, [m][k])
    int o = i >> 6, c = i & 63;
    Ah[o * PITCH + c] = (_Float16)Wt[i];
  }
  for (int i = tid; i < 64 * 64; i += 256) {          // X tile -> LDS ([n][k])
    int n = i & 63, c = i >> 6;
    Bt[n * PITCH + c] = (_Float16)Xb[(long)c * Ntot + n_base + n];
  }
  __syncthreads();

  const int w = tid >> 5, lane = tid & 31;
  const int m0 = (w >> 1) * 16, n0 = (w & 1) * 32;
  const int hi = lane >> 4, ln = lane & 15;

  v8f acc0 = {}, acc1 = {};
#pragma unroll
  for (int ks = 0; ks < 2; ++ks) {
    v16h a  = frag_a(Ah, m0, ks * 32, lane);
    v16h b0 = frag_b(Bt, n0,      ks * 32, lane);
    v16h b1 = frag_b(Bt, n0 + 16, ks * 32, lane);
    acc0 = __builtin_amdgcn_wmma_f32_16x16x32_f16(false, a, false, b0,
                                                  (short)0, acc0, false, false);
    acc1 = __builtin_amdgcn_wmma_f32_16x16x32_f16(false, a, false, b1,
                                                  (short)0, acc1, false, false);
  }

#pragma unroll
  for (int t = 0; t < 2; ++t) {
    const v8f& acc = t ? acc1 : acc0;
    int gn = n_base + n0 + t * 16 + ln;
#pragma unroll
    for (int j = 0; j < 8; ++j) {
      int gm = m0 + j + 8 * hi;
      float v = acc[j] + bias[gm];
      if (mode == 0)      outF[((long)b * 64 + gm) * Ntot + gn] = v;
      else if (mode == 1) outH[((long)b * 64 + gm) * Ntot + gn] = (_Float16)v;
      else                outH[((long)b * Ntot + gn) * 64 + gm] = (_Float16)v;
    }
  }
}

// ---------------------------------------------------------------------------
// Kernel 2: 2x2 average pool in the (C, W, H) reshaped view.
// q f32 (B,64,4096) -> q_h (B,64,1024) f16 and qT_h (B,1024,64) f16
// ---------------------------------------------------------------------------
__global__ void pool_kernel(const float* __restrict__ q,
                            _Float16* __restrict__ qh,
                            _Float16* __restrict__ qTh) {
  int idx = blockIdx.x * 256 + threadIdx.x;     // B*64*1024 total
  int p = idx & 1023;
  int c = (idx >> 10) & 63;
  int b = idx >> 16;
  int w2 = p >> 5, h2 = p & 31;
  const float* qb = q + ((long)b * 64 + c) * 4096;
  int i0 = (2 * w2) * 64 + 2 * h2;
  float v = 0.25f * (qb[i0] + qb[i0 + 1] + qb[i0 + 64] + qb[i0 + 65]);
  qh[((long)b * 64 + c) * 1024 + p] = (_Float16)v;
  qTh[((long)b * 1024 + p) * 64 + c] = (_Float16)v;
}

// ---------------------------------------------------------------------------
// Kernel 3: logits GEMM  S(MxN) = A(Mx64) * B(64xN), f16, K=64 fixed.
// A row-major lda=64, B row-major ldb, S row-major ldS.
// grid: (N/64, M/64, B), block 256.
// ---------------------------------------------------------------------------
__global__ void logits_gemm(const _Float16* __restrict__ A,
                            const _Float16* __restrict__ B,
                            _Float16* __restrict__ S,
                            int ldb, int ldS,
                            long aBatch, long bBatch, long sBatch) {
  __shared__ __align__(16) _Float16 Ah[64 * PITCH];
  __shared__ __align__(16) _Float16 Bt[64 * PITCH];
  const int b = blockIdx.z;
  const _Float16* Ab = A + (long)b * aBatch;
  const _Float16* Bb = B + (long)b * bBatch;
  _Float16* Sb = S + (long)b * sBatch;
  const int m_base = blockIdx.y * 64;
  const int n_base = blockIdx.x * 64;
  const int tid = threadIdx.x;

  for (int i = tid; i < 64 * 64; i += 256) {          // A -> LDS ([m][k])
    int m = i >> 6, k = i & 63;
    Ah[m * PITCH + k] = Ab[(long)(m_base + m) * 64 + k];
  }
  for (int i = tid; i < 64 * 64; i += 256) {          // B -> LDS ([n][k])
    int n = i & 63, k = i >> 6;
    Bt[n * PITCH + k] = Bb[(long)k * ldb + n_base + n];
  }
  __syncthreads();

  const int w = tid >> 5, lane = tid & 31;
  const int m0 = (w >> 1) * 16, n0 = (w & 1) * 32;
  const int hi = lane >> 4, ln = lane & 15;

  v8f acc0 = {}, acc1 = {};
#pragma unroll
  for (int ks = 0; ks < 2; ++ks) {
    v16h a  = frag_a(Ah, m0, ks * 32, lane);
    v16h b0 = frag_b(Bt, n0,      ks * 32, lane);
    v16h b1 = frag_b(Bt, n0 + 16, ks * 32, lane);
    acc0 = __builtin_amdgcn_wmma_f32_16x16x32_f16(false, a, false, b0,
                                                  (short)0, acc0, false, false);
    acc1 = __builtin_amdgcn_wmma_f32_16x16x32_f16(false, a, false, b1,
                                                  (short)0, acc1, false, false);
  }

#pragma unroll
  for (int t = 0; t < 2; ++t) {
    const v8f& acc = t ? acc1 : acc0;
    int gn = n_base + n0 + t * 16 + ln;
#pragma unroll
    for (int j = 0; j < 8; ++j) {
      int gm = m_base + m0 + j + 8 * hi;
      Sb[(long)gm * ldS + gn] = (_Float16)acc[j];
    }
  }
}

// ---------------------------------------------------------------------------
// Kernel 4: row softmax in place over f16 matrix; one block per row.
// ---------------------------------------------------------------------------
__global__ void softmax_kernel(_Float16* __restrict__ attn, int cols) {
  long row = blockIdx.x;
  _Float16* r = attn + row * (long)cols;
  __shared__ float red[256];
  const int tid = threadIdx.x;

  float mx = -1e30f;
  for (int i = tid; i < cols; i += 256) mx = fmaxf(mx, (float)r[i]);
  red[tid] = mx;
  __syncthreads();
  for (int s = 128; s > 0; s >>= 1) {
    if (tid < s) red[tid] = fmaxf(red[tid], red[tid + s]);
    __syncthreads();
  }
  mx = red[0];
  __syncthreads();

  float sum = 0.f;
  for (int i = tid; i < cols; i += 256) sum += __expf((float)r[i] - mx);
  red[tid] = sum;
  __syncthreads();
  for (int s = 128; s > 0; s >>= 1) {
    if (tid < s) red[tid] += red[tid + s];
    __syncthreads();
  }
  float inv = 1.0f / red[0];

  for (int i = tid; i < cols; i += 256)
    r[i] = (_Float16)(__expf((float)r[i] - mx) * inv);
}

// ---------------------------------------------------------------------------
// Kernel 5: M=64 GEMM  D(64xN) = A(64xK) * B(KxN)
//   A f16 row-major (lda); B element (k, j) at Bp[j*ldbj + k] (k contiguous,
//   i.e. B = attn^T -> LDS fill is fully coalesced AND conflict-free).
//   K looped in 64-wide LDS chunks with global prefetch of the next chunk.
// mode 0: f16 out1[gm*ldo + gn] | mode 1: f32 d_out = gamma*acc + x (residual)
// grid: (N/64, B), block 256.
// ---------------------------------------------------------------------------
__global__ void gemm_m64(const _Float16* __restrict__ A,
                         const _Float16* __restrict__ Bp,
                         _Float16* __restrict__ outH, float* __restrict__ outF,
                         const float* __restrict__ X,
                         const float* __restrict__ gamma,
                         int K, int Nn, int lda, int ldbj, int ldo, int mode,
                         long aBatch, long bBatch, long oBatch) {
  __shared__ __align__(16) _Float16 Ah[64 * PITCH];
  __shared__ __align__(16) _Float16 Bt[64 * PITCH];
  const int b = blockIdx.y;
  const _Float16* Ab = A + (long)b * aBatch;
  const _Float16* Bb = Bp + (long)b * bBatch;
  const int n_base = blockIdx.x * 64;
  const int tid = threadIdx.x;

  const int w = tid >> 5, lane = tid & 31;
  const int m0 = (w >> 1) * 16, n0 = (w & 1) * 32;
  const int hi = lane >> 4, ln = lane & 15;

  v8f acc0 = {}, acc1 = {};
  for (int k0 = 0; k0 < K; k0 += 64) {
    __syncthreads();
    for (int i = tid; i < 64 * 64; i += 256) {      // A chunk [m][k]
      int k = i & 63, m = i >> 6;
      Ah[m * PITCH + k] = Ab[(long)m * lda + k0 + k];
    }
    for (int i = tid; i < 64 * 64; i += 256) {      // B chunk [j][k] (natural!)
      int k = i & 63, j = i >> 6;
      Bt[j * PITCH + k] = Bb[(long)(n_base + j) * ldbj + k0 + k];
    }
    if (k0 + 64 < K) {                              // prefetch next chunks
      __builtin_prefetch(&Ab[(long)(tid >> 2) * lda + k0 + 64], 0, 1);
      __builtin_prefetch(&Bb[(long)(n_base + (tid >> 2)) * ldbj + k0 + 64], 0, 1);
    }
    __syncthreads();
#pragma unroll
    for (int ks = 0; ks < 2; ++ks) {
      v16h a  = frag_a(Ah, m0, ks * 32, lane);
      v16h b0 = frag_b(Bt, n0,      ks * 32, lane);
      v16h b1 = frag_b(Bt, n0 + 16, ks * 32, lane);
      acc0 = __builtin_amdgcn_wmma_f32_16x16x32_f16(false, a, false, b0,
                                                    (short)0, acc0, false, false);
      acc1 = __builtin_amdgcn_wmma_f32_16x16x32_f16(false, a, false, b1,
                                                    (short)0, acc1, false, false);
    }
  }

#pragma unroll
  for (int t = 0; t < 2; ++t) {
    const v8f& acc = t ? acc1 : acc0;
    int gn = n_base + n0 + t * 16 + ln;
    if (mode == 0) {
      _Float16* Ob = outH + (long)b * oBatch;
#pragma unroll
      for (int j = 0; j < 8; ++j) {
        int gm = m0 + j + 8 * hi;
        Ob[(long)gm * ldo + gn] = (_Float16)acc[j];
      }
    } else {
      float g = gamma[0];
#pragma unroll
      for (int j = 0; j < 8; ++j) {
        int gm = m0 + j + 8 * hi;
        long idx = ((long)b * 64 + gm) * (long)Nn + gn;
        outF[idx] = g * acc[j] + X[idx];
      }
    }
  }
}

// ---------------------------------------------------------------------------
// Host launcher
// ---------------------------------------------------------------------------
extern "C" void kernel_launch(void* const* d_in, const int* in_sizes, int n_in,
                              void* d_out, int out_size, void* d_ws, size_t ws_size,
                              hipStream_t stream) {
  (void)in_sizes; (void)n_in; (void)out_size; (void)ws_size;
  const float* x     = (const float*)d_in[0];
  const float* wq    = (const float*)d_in[1];
  const float* bq    = (const float*)d_in[2];
  const float* wQ    = (const float*)d_in[3];
  const float* bQ    = (const float*)d_in[4];
  const float* wK    = (const float*)d_in[5];
  const float* bK    = (const float*)d_in[6];
  const float* wV    = (const float*)d_in[7];
  const float* bV    = (const float*)d_in[8];
  const float* gamma = (const float*)d_in[9];

  const int B = 8, C = 64, N = 4096, P = 1024;

  char* ws = (char*)d_ws;
  size_t off = 0;
  auto alloc = [&](size_t bytes) -> void* {
    void* p = ws + off;
    off += (bytes + 255) & ~(size_t)255;
    return p;
  };
  float*    q_f32  = (float*)   alloc((size_t)B * C * N * 4);
  _Float16* q_h    = (_Float16*)alloc((size_t)B * C * P * 2);
  _Float16* qT_h   = (_Float16*)alloc((size_t)B * P * C * 2);
  _Float16* K_h    = (_Float16*)alloc((size_t)B * C * N * 2);
  _Float16* V_h    = (_Float16*)alloc((size_t)B * C * N * 2);
  _Float16* QT_h   = (_Float16*)alloc((size_t)B * N * C * 2);
  _Float16* out1_h = (_Float16*)alloc((size_t)B * C * P * 2);
  _Float16* attn   = (_Float16*)alloc((size_t)B * P * N * 2);  // reused for attn2

  // 1x1 convolutions
  conv_gemm<<<dim3(N / 64, B), 256, 0, stream>>>(x, wq, bq, q_f32, nullptr, 0, N);
  conv_gemm<<<dim3(N / 64, B), 256, 0, stream>>>(x, wK, bK, nullptr, K_h, 1, N);
  conv_gemm<<<dim3(N / 64, B), 256, 0, stream>>>(x, wV, bV, nullptr, V_h, 1, N);
  conv_gemm<<<dim3(N / 64, B), 256, 0, stream>>>(x, wQ, bQ, nullptr, QT_h, 2, N);

  // 2x2 average pool -> q_ in both layouts
  pool_kernel<<<(B * C * P) / 256, 256, 0, stream>>>(q_f32, q_h, qT_h);

  // stage 1: attn1 = softmax_n(q_^T K)   (B, 1024, 4096)
  logits_gemm<<<dim3(N / 64, P / 64, B), 256, 0, stream>>>(
      qT_h, K_h, attn, /*ldb=*/N, /*ldS=*/N,
      (long)P * C, (long)C * N, (long)P * N);
  softmax_kernel<<<B * P, 256, 0, stream>>>(attn, N);

  // out1 = V * attn1^T   (B, 64, 1024)
  gemm_m64<<<dim3(P / 64, B), 256, 0, stream>>>(
      V_h, attn, out1_h, nullptr, nullptr, nullptr,
      /*K=*/N, /*Nn=*/P, /*lda=*/N, /*ldbj=*/N, /*ldo=*/P, /*mode=*/0,
      (long)C * N, (long)P * N, (long)C * P);

  // stage 2: attn2 = softmax_p(Q^T q_)   (B, 4096, 1024)
  logits_gemm<<<dim3(P / 64, N / 64, B), 256, 0, stream>>>(
      QT_h, q_h, attn, /*ldb=*/P, /*ldS=*/P,
      (long)N * C, (long)C * P, (long)N * P);
  softmax_kernel<<<B * N, 256, 0, stream>>>(attn, P);

  // final = gamma * (out1 * attn2^T) + x   -> d_out
  gemm_m64<<<dim3(N / 64, B), 256, 0, stream>>>(
      out1_h, attn, nullptr, (float*)d_out, x, gamma,
      /*K=*/P, /*Nn=*/N, /*lda=*/P, /*ldbj=*/P, /*ldo=*/0, /*mode=*/1,
      (long)C * P, (long)N * P, (long)0);
}
